// LinearAttentionBase_30013231464543
// MI455X (gfx1250) — matrix operand order repaired and checked by reading, not compiled, not run on hardware
//
#include <hip/hip_runtime.h>

// ---- types ------------------------------------------------------------
typedef __attribute__((ext_vector_type(16))) __bf16       v16bf;
typedef __attribute__((ext_vector_type(8)))  float        v8f;
typedef __attribute__((ext_vector_type(4)))  unsigned int u32x4;

union Pack8 { __bf16 b[8]; u32x4 u; };        // 16 bytes, trivial
union Frag  { u32x4 u[2]; v16bf v; };         // 32 bytes, trivial

#define B_    4
#define H_    16
#define S_    4096
#define D_    128
#define BH_   (B_ * H_)
#define EPS_  1e-6f
#define HEAD_WS (D_ * D_ + D_)               // KV (128x128) + ksum (128) floats
#define SPLIT_ 8                             // S-split factor for phase 1

__device__ __forceinline__ float featmap(float x) {
    // elu(x) + 1 == (x > 0) ? x + 1 : exp(x)
    return x > 0.0f ? (x + 1.0f) : __expf(x);
}

// =======================================================================
// Phase 1: partial KV = kf^T * v over an S-slice, partial k_sum.
// grid = (nsplit, BH_), 512 threads (16 waves); wave owns 4 16x16 tiles.
// S-slice processed in CHUNKS chunks of 64 (two K=32 WMMA sub-steps each).
// CHUNKS is a compile-time constant so global addressing strength-reduces
// to base + immediate-offset load clauses.
// =======================================================================
template <int CHUNKS>
__global__ __launch_bounds__(512)
void la_phase1(const float* __restrict__ k, const float* __restrict__ v,
               float* __restrict__ dst) {
    __shared__ __align__(16) __bf16 kfT[D_ * 64];   // [d][s]  16 KB
    __shared__ __align__(16) __bf16 vT [D_ * 64];   // [e][s]  16 KB
    __shared__ float red[512];

    const int  sp    = blockIdx.x;             // split index
    const int  head  = blockIdx.y;
    const long base  = (long)head * S_ * D_;
    const int  sbeg  = sp * CHUNKS * 64;       // first s of this slice
    const int  tid   = threadIdx.x;
    const int  lane  = tid & 31;
    const int  wave  = tid >> 5;
    const int  m0    = (wave >> 1) * 16;       // 8 row-strips
    const int  n0    = (wave & 1) * 64;        // 2 col-halves, 4 tiles each
    const int  l15   = lane & 15;
    const int  half  = (lane >> 4) & 1;

    const int  d     = tid & 127;              // column owned during convert
    const int  sh    = tid >> 7;               // 16-row group (0..3)

    const v8f zero8 = {0.f,0.f,0.f,0.f,0.f,0.f,0.f,0.f};
    v8f acc[4] = {zero8, zero8, zero8, zero8};
    float ksum_acc = 0.0f;

    const char* kfTb = (const char*)kfT;
    const char* vTb  = (const char*)vT;
    const int arow = m0 + l15;

    for (int c = 0; c < CHUNKS; ++c) {
        const int s0 = sbeg + c * 64;

        // ---- load + feature map + bf16 convert (lane-coalesced rows) --
        Pack8 pk[2], pv[2];
#pragma unroll
        for (int j = 0; j < 2; ++j) {
#pragma unroll
            for (int i = 0; i < 8; ++i) {
                const int  s  = sh * 16 + j * 8 + i;
                const long gi = base + (long)(s0 + s) * D_ + d;
                const float fk = featmap(k[gi]);
                ksum_acc += fk;
                pk[j].b[i] = (__bf16)fk;
                pv[j].b[i] = (__bf16)v[gi];
            }
        }
        // prefetch next chunk while this one is staged & consumed
        if (c + 1 < CHUNKS) {
            const long pg = base + (long)(s0 + 64 + sh * 16) * D_ + d;
            __builtin_prefetch(&k[pg]);
            __builtin_prefetch(&v[pg]);
        }
        __syncthreads();   // previous iteration's readers are done
#pragma unroll
        for (int j = 0; j < 2; ++j) {
            *(u32x4*)((char*)kfT + d * 128 + sh * 32 + j * 16) = pk[j].u;
            *(u32x4*)((char*)vT  + d * 128 + sh * 32 + j * 16) = pv[j].u;
        }
        __syncthreads();

        // ---- WMMA: A = kf^T tile (16x32 x2), B = v tile (32x16 x2) ----
        Frag af0, af1;
        af0.u[0] = *(const u32x4*)(kfTb + arow * 128 + half * 16);
        af0.u[1] = *(const u32x4*)(kfTb + arow * 128 + 32 + half * 16);
        af1.u[0] = *(const u32x4*)(kfTb + arow * 128 + 64 + half * 16);
        af1.u[1] = *(const u32x4*)(kfTb + arow * 128 + 96 + half * 16);
#pragma unroll
        for (int t = 0; t < 4; ++t) {
            const int e = n0 + t * 16 + l15;
            Frag b0, b1;
            b0.u[0] = *(const u32x4*)(vTb + e * 128 + half * 16);
            b0.u[1] = *(const u32x4*)(vTb + e * 128 + 32 + half * 16);
            b1.u[0] = *(const u32x4*)(vTb + e * 128 + 64 + half * 16);
            b1.u[1] = *(const u32x4*)(vTb + e * 128 + 96 + half * 16);
            acc[t] = __builtin_amdgcn_wmma_f32_16x16x32_bf16(
                false, af0.v, false, b0.v, (short)0, acc[t], false, false);
            acc[t] = __builtin_amdgcn_wmma_f32_16x16x32_bf16(
                false, af1.v, false, b1.v, (short)0, acc[t], false, false);
        }
    }

    // ---- k_sum reduction (4 partials per d) ---------------------------
    red[tid] = ksum_acc;
    __syncthreads();
    float* wsh = dst + (long)(head * gridDim.x + sp) * HEAD_WS;
    if (tid < 128) {
        wsh[D_ * D_ + tid] = red[tid] + red[tid + 128] + red[tid + 256] + red[tid + 384];
    }

    // ---- write partial KV (f32) --------------------------------------
#pragma unroll
    for (int t = 0; t < 4; ++t) {
        const int col = n0 + t * 16 + l15;
#pragma unroll
        for (int r = 0; r < 8; ++r) {
            const int row = m0 + half * 8 + r;
            wsh[row * D_ + col] = acc[t][r];
        }
    }
}

// =======================================================================
// Reduce SPLIT_ partial (KV, ksum) slabs into the final per-head slab.
// Fixed summation order -> deterministic. grid = BH_, 512 threads.
// =======================================================================
__global__ __launch_bounds__(512)
void la_reduce(const float* __restrict__ part, float* __restrict__ fin) {
    const int head = blockIdx.x;
    const int tid  = threadIdx.x;
    const float* p0 = part + (long)head * SPLIT_ * HEAD_WS;
    float*       f0 = fin  + (long)head * HEAD_WS;
    for (int e = tid; e < HEAD_WS; e += 512) {
        float s = 0.0f;
#pragma unroll
        for (int sp = 0; sp < SPLIT_; ++sp)
            s += p0[(long)sp * HEAD_WS + e];
        f0[e] = s;
    }
}

// =======================================================================
// Phase 2: out = (qf * KV) / (qf . k_sum + eps)
// grid = BH_*32 blocks (128 q-rows each), 512 threads (16 waves).
// D=128 reduced in 2 chunks of 64 (two K=32 WMMA sub-steps each).
// =======================================================================
__global__ __launch_bounds__(512)
void la_phase2(const float* __restrict__ q, const float* __restrict__ ws,
               float* __restrict__ out) {
    __shared__ __align__(16) __bf16 KVT[D_ * D_];   // [e][d] bf16, 32 KB
    __shared__ __align__(16) __bf16 qA [D_ * 64];   // [row][d-chunk], 16 KB
    __shared__ float ksumL[D_];
    __shared__ float red[512];
    __shared__ float deninv[D_];

    const int  head  = blockIdx.x >> 5;
    const int  sblk  = blockIdx.x & 31;
    const int  sbase = sblk * 128;
    const long qbase = (long)head * S_ * D_;
    const float* wsh = ws + (long)head * HEAD_WS;

    const int tid  = threadIdx.x;
    const int lane = tid & 31;
    const int wave = tid >> 5;
    const int m0   = (wave >> 1) * 16;
    const int n0   = (wave & 1) * 64;
    const int l15  = lane & 15;
    const int half = (lane >> 4) & 1;

    // ---- stage 0: KV -> bf16 transposed LDS, ksum -> LDS --------------
    {
        const int e  = tid & 127;
        const int db = (tid >> 7) * 32;
#pragma unroll
        for (int j0 = 0; j0 < 32; j0 += 8) {
            Pack8 p;
#pragma unroll
            for (int i = 0; i < 8; ++i)
                p.b[i] = (__bf16)wsh[(db + j0 + i) * D_ + e];
            *(u32x4*)((char*)KVT + e * 256 + (db + j0) * 2) = p.u;
        }
        if (tid < 128) ksumL[tid] = wsh[D_ * D_ + tid];
    }
    __syncthreads();

    const int row = tid & 127;
    const int dh  = tid >> 7;
    float den_acc = 0.0f;

    const v8f zero8 = {0.f,0.f,0.f,0.f,0.f,0.f,0.f,0.f};
    v8f acc[4] = {zero8, zero8, zero8, zero8};

    const char* qAb  = (const char*)qA;
    const char* KVTb = (const char*)KVT;
    const int arow = m0 + l15;

#pragma unroll
    for (int cc = 0; cc < 2; ++cc) {
        const int d0 = cc * 64;

        Pack8 pq[2];
#pragma unroll
        for (int j = 0; j < 2; ++j) {
#pragma unroll
            for (int i = 0; i < 8; ++i) {
                const int dd = d0 + dh * 16 + j * 8 + i;
                const float f = featmap(q[qbase + (long)(sbase + row) * D_ + dd]);
                den_acc += f * ksumL[dd];        // exact f32 denominator
                pq[j].b[i] = (__bf16)f;
            }
        }
        __syncthreads();
#pragma unroll
        for (int j = 0; j < 2; ++j)
            *(u32x4*)((char*)qA + row * 128 + dh * 32 + j * 16) = pq[j].u;
        __syncthreads();

        Frag af0, af1;
        af0.u[0] = *(const u32x4*)(qAb + arow * 128 + half * 16);
        af0.u[1] = *(const u32x4*)(qAb + arow * 128 + 32 + half * 16);
        af1.u[0] = *(const u32x4*)(qAb + arow * 128 + 64 + half * 16);
        af1.u[1] = *(const u32x4*)(qAb + arow * 128 + 96 + half * 16);
#pragma unroll
        for (int t = 0; t < 4; ++t) {
            const int e = n0 + t * 16 + l15;
            Frag b0, b1;
            b0.u[0] = *(const u32x4*)(KVTb + e * 256 + d0 * 2 + half * 16);
            b0.u[1] = *(const u32x4*)(KVTb + e * 256 + d0 * 2 + 32 + half * 16);
            b1.u[0] = *(const u32x4*)(KVTb + e * 256 + d0 * 2 + 64 + half * 16);
            b1.u[1] = *(const u32x4*)(KVTb + e * 256 + d0 * 2 + 96 + half * 16);
            acc[t] = __builtin_amdgcn_wmma_f32_16x16x32_bf16(
                false, af0.v, false, b0.v, (short)0, acc[t], false, false);
            acc[t] = __builtin_amdgcn_wmma_f32_16x16x32_bf16(
                false, af1.v, false, b1.v, (short)0, acc[t], false, false);
        }
    }

    // ---- denominator reduce ------------------------------------------
    red[tid] = den_acc;
    __syncthreads();
    if (tid < 128) {
        const float den = red[tid] + red[tid + 128] + red[tid + 256] + red[tid + 384];
        deninv[tid] = 1.0f / (den + EPS_);
    }
    __syncthreads();

    // ---- scale + store ------------------------------------------------
#pragma unroll
    for (int t = 0; t < 4; ++t) {
        const int col = n0 + t * 16 + l15;
#pragma unroll
        for (int r = 0; r < 8; ++r) {
            const int rw = m0 + half * 8 + r;
            out[qbase + (long)(sbase + rw) * D_ + col] = acc[t][r] * deninv[rw];
        }
    }
}

// =======================================================================
extern "C" void kernel_launch(void* const* d_in, const int* in_sizes, int n_in,
                              void* d_out, int out_size, void* d_ws, size_t ws_size,
                              hipStream_t stream) {
    const float* q = (const float*)d_in[0];
    const float* k = (const float*)d_in[1];
    const float* v = (const float*)d_in[2];
    float* out = (float*)d_out;
    float* ws  = (float*)d_ws;

    // Layout: [final: BH_*HEAD_WS][partials: BH_*SPLIT_*HEAD_WS]
    const size_t final_bytes = (size_t)BH_ * HEAD_WS * sizeof(float);      // ~4.0 MB
    const size_t split_bytes = final_bytes * (size_t)(1 + SPLIT_);         // ~36.3 MB

    if (ws_size >= split_bytes) {
        // High-parallelism path: 512 phase-1 blocks + deterministic reduce.
        float* part = ws + (size_t)BH_ * HEAD_WS;
        la_phase1<S_ / (64 * SPLIT_)><<<dim3(SPLIT_, BH_), dim3(512), 0, stream>>>(k, v, part);
        la_reduce<<<dim3(BH_), dim3(512), 0, stream>>>(part, ws);
    } else {
        // Fallback: one block per head writes the final slab directly.
        la_phase1<S_ / 64><<<dim3(1, BH_), dim3(512), 0, stream>>>(k, v, ws);
    }
    la_phase2<<<dim3(BH_ * 32), dim3(512), 0, stream>>>(q, ws, out);
}
